// DCN_69887707840547
// MI455X (gfx1250) — matrix operand order repaired and checked by reading, not compiled, not run on hardware
//
#include <hip/hip_runtime.h>
#include <math.h>

typedef __attribute__((ext_vector_type(16))) _Float16 v16h;
typedef __attribute__((ext_vector_type(8)))  _Float16 v8h;
typedef __attribute__((ext_vector_type(8)))  float    v8f;

#define Bn   8
#define CINC 64
#define Hh   128
#define Wd   128
#define HW   (Hh*Wd)
#define COUTC 128
#define KKC  9

static __device__ __forceinline__ v16h mk16(v8h lo, v8h hi) {
  v16h r;
#pragma unroll
  for (int i = 0; i < 8; ++i) { r[i] = lo[i]; r[i + 8] = hi[i]; }
  return r;
}

// ---------------------------------------------------------------------------
// Kernel 0: weight prep. pre_w -> f16 [co][ci]; reg_w -> f16 [kk][co][ci]
// ---------------------------------------------------------------------------
__global__ void k_prep(const float* __restrict__ pre_w,
                       const float* __restrict__ reg_w,
                       _Float16* __restrict__ wpreh,
                       _Float16* __restrict__ wregh) {
  int i = blockIdx.x * 256 + threadIdx.x;
  if (i < CINC * CINC) wpreh[i] = (_Float16)pre_w[i];
  int j = i - CINC * CINC;
  if (j >= 0 && j < KKC * COUTC * CINC) {
    int kk = j / (COUTC * CINC);
    int r  = j % (COUTC * CINC);
    int co = r / CINC, ci = r % CINC;
    wregh[j] = (_Float16)reg_w[(co * CINC + ci) * KKC + kk];
  }
}

// ---------------------------------------------------------------------------
// Kernel 1: 1x1 pre-conv as WMMA GEMM.  M=64 couts, K=64 cins, N=64 pixels/blk
// Block: 256 thr (8 waves). Waves: m0=(w&3)*16, n-half=(w>>2), 2 accums each.
// ---------------------------------------------------------------------------
__global__ void __launch_bounds__(256)
k_preconv(const float* __restrict__ x, const _Float16* __restrict__ wpreh,
          const float* __restrict__ pre_b, float* __restrict__ xp) {
  __shared__ __align__(16) _Float16 Xt[64 * 72];  // [pixel][ci], K-contiguous
  const int t    = threadIdx.x;
  const int bid  = blockIdx.x;
  const int b    = bid / (HW / 64);
  const int pix0 = (bid % (HW / 64)) * 64;
  const float* xb = x + (size_t)b * CINC * HW;

  // stage 64ci x 64p, transpose to [p][ci] f16 (coalesced global reads)
#pragma unroll
  for (int i = 0; i < 16; ++i) {
    int flat = t + 256 * i;          // 0..4095
    int ci = flat >> 6, p = flat & 63;
    Xt[p * 72 + ci] = (_Float16)xb[(size_t)ci * HW + pix0 + p];
  }
  __syncthreads();

  const int w = t >> 5, lane = t & 31;
  const int m0    = (w & 3) * 16;
  const int nbase = (w >> 2) * 32;
  const int ml = lane & 15, sel = lane >> 4;
  v8f acc0 = {}; v8f acc1 = {};

#pragma unroll
  for (int c0 = 0; c0 < CINC; c0 += 32) {
    // A: lanes<16 K0..15 contiguous; lanes>=16 K16..31
    const _Float16* ap = wpreh + (m0 + ml) * CINC + c0 + sel * 16;
    v16h a = mk16(*(const v8h*)ap, *(const v8h*)(ap + 8));
    // B: lanes<16 K0..15 contiguous; lanes>=16 K16..31
    const int kb = sel * 16;
    const _Float16* bp0 = &Xt[(nbase + ml) * 72 + c0 + kb];
    v16h b0 = mk16(*(const v8h*)bp0, *(const v8h*)(bp0 + 8));
    const _Float16* bp1 = &Xt[(nbase + 16 + ml) * 72 + c0 + kb];
    v16h b1 = mk16(*(const v8h*)bp1, *(const v8h*)(bp1 + 8));
    acc0 = __builtin_amdgcn_wmma_f32_16x16x32_f16(false, a, false, b0, (short)0, acc0, false, false);
    acc1 = __builtin_amdgcn_wmma_f32_16x16x32_f16(false, a, false, b1, (short)0, acc1, false, false);
  }

  float* xpb = xp + (size_t)b * CINC * HW;
#pragma unroll
  for (int v = 0; v < 8; ++v) {
    int m = m0 + v + 8 * sel;
    float bias = pre_b[m];
    int n = lane & 15;
    xpb[(size_t)m * HW + pix0 + nbase + n]      = acc0[v] + bias;
    xpb[(size_t)m * HW + pix0 + nbase + 16 + n] = acc1[v] + bias;
  }
}

// ---------------------------------------------------------------------------
// Kernel 2: offset (18ch) + modulation (9ch) 3x3 convs, direct, weights in LDS.
// Block = 16x16 pixel tile, 1 thread per pixel, 27 accumulators in regs.
// ---------------------------------------------------------------------------
__global__ void __launch_bounds__(256)
k_offmask(const float* __restrict__ xp,
          const float* __restrict__ off_w, const float* __restrict__ off_b,
          const float* __restrict__ mod_w, const float* __restrict__ mod_b,
          float* __restrict__ offs, float* __restrict__ maskb) {
  __shared__ float wl[27 * 576];  // [co][ci][tap], co<18 = off_w, else mod_w
  const int t = threadIdx.x;
  for (int i = t; i < 27 * 576; i += 256)
    wl[i] = (i < 18 * 576) ? off_w[i] : mod_w[i - 18 * 576];

  const int bid = blockIdx.x;
  const int b = bid / 64, tile = bid % 64;
  const int oy = (tile >> 3) * 16 + (t >> 4);
  const int ox = (tile & 7) * 16 + (t & 15);
  __syncthreads();

  float acc[27];
#pragma unroll
  for (int co = 0; co < 27; ++co) acc[co] = (co < 18) ? off_b[co] : mod_b[co - 18];

  const float* xpb = xp + (size_t)b * CINC * HW;
  for (int ci = 0; ci < CINC; ++ci) {
    float tap[9];
#pragma unroll
    for (int ky = 0; ky < 3; ++ky)
#pragma unroll
      for (int kx = 0; kx < 3; ++kx) {
        int yy = oy - 1 + ky, xx = ox - 1 + kx;
        float v = 0.f;
        if ((unsigned)yy < (unsigned)Hh && (unsigned)xx < (unsigned)Wd)
          v = xpb[(size_t)ci * HW + yy * Wd + xx];
        tap[ky * 3 + kx] = v;
      }
#pragma unroll
    for (int co = 0; co < 27; ++co) {
      float a = acc[co];
#pragma unroll
      for (int tp = 0; tp < 9; ++tp)
        a = fmaf(wl[co * 576 + ci * 9 + tp], tap[tp], a);
      acc[co] = a;
    }
  }

  const int p = oy * Wd + ox;
#pragma unroll
  for (int co = 0; co < 18; ++co)
    offs[((size_t)b * 18 + co) * HW + p] = acc[co];
#pragma unroll
  for (int co = 0; co < 9; ++co)
    maskb[((size_t)b * 9 + co) * HW + p] = 2.f / (1.f + __expf(-acc[18 + co]));
}

// ---------------------------------------------------------------------------
// Kernel 3: fused modulated-bilinear-sampling + final GEMM (WMMA).
// Per block: 64 pixels x 128 couts. K-loop over kk(9) x cin-chunk(2 x 32).
// B-tile sampled on the fly into LDS (f16, K-contiguous per column).
// Each of 8 waves: m0 = wave*16, 4 n-tiles, 4 f32 accumulators.
// ---------------------------------------------------------------------------
__global__ void __launch_bounds__(256)
k_dcn_main(const float* __restrict__ xp, const float* __restrict__ offs,
           const float* __restrict__ maskb, const _Float16* __restrict__ wregh,
           float* __restrict__ out) {
  __shared__ __align__(16) _Float16 Bt[64 * 40];  // [pixel][ci], stride 40 halves
  __shared__ int   sY[64], sX[64];
  __shared__ float sW00[64], sW01[64], sW10[64], sW11[64], sM[64];

  const int t    = threadIdx.x;
  const int bid  = blockIdx.x;
  const int b    = bid / (HW / 64);
  const int pix0 = (bid % (HW / 64)) * 64;
  const int w = t >> 5, lane = t & 31;
  const int m0 = w * 16;
  const int ml = lane & 15, sel = lane >> 4;
  const float* xpb = xp + (size_t)b * CINC * HW;

  v8f acc[4] = {};

  for (int kk = 0; kk < KKC; ++kk) {
    __syncthreads();
    if (t < 64) {  // per-pixel sampling coords for this kernel tap
      int p = pix0 + t;
      int y = p >> 7, x = p & 127;
      float dy = offs[((size_t)b * 18 + 2 * kk) * HW + p];
      float dx = offs[((size_t)b * 18 + 2 * kk + 1) * HW + p];
      float m  = maskb[((size_t)b * 9 + kk) * HW + p];
      float py = (float)(y - 1 + kk / 3) + dy;
      float px = (float)(x - 1 + kk % 3) + dx;
      float fy = floorf(py), fx = floorf(px);
      float wy = py - fy, wx = px - fx;
      sY[t] = (int)fy; sX[t] = (int)fx;
      sW00[t] = (1.f - wy) * (1.f - wx);
      sW01[t] = (1.f - wy) * wx;
      sW10[t] = wy * (1.f - wx);
      sW11[t] = wy * wx;
      sM[t] = m;
    }
    __syncthreads();

    for (int c0 = 0; c0 < CINC; c0 += 32) {
      {  // stage B tile: 32 ci x 64 pixels, 8 samples per thread
        const int pl = t & 63;
        const int cb = (t >> 6) * 8;
        int y0 = sY[pl], x0 = sX[pl];
        float w00 = sW00[pl], w01 = sW01[pl], w10 = sW10[pl], w11 = sW11[pl];
        float m = sM[pl];
        int y1 = y0 + 1, x1 = x0 + 1;
        bool v00 = ((unsigned)y0 < (unsigned)Hh) & ((unsigned)x0 < (unsigned)Wd);
        bool v01 = ((unsigned)y0 < (unsigned)Hh) & ((unsigned)x1 < (unsigned)Wd);
        bool v10 = ((unsigned)y1 < (unsigned)Hh) & ((unsigned)x0 < (unsigned)Wd);
        bool v11 = ((unsigned)y1 < (unsigned)Hh) & ((unsigned)x1 < (unsigned)Wd);
        int i00 = y0 * Wd + x0, i01 = y0 * Wd + x1;
        int i10 = y1 * Wd + x0, i11 = y1 * Wd + x1;
        const float* base = xpb + (size_t)(c0 + cb) * HW;
#pragma unroll
        for (int j = 0; j < 8; ++j) {
          const float* r = base + (size_t)j * HW;
          float v = 0.f;
          if (v00) v = fmaf(w00, r[i00], v);
          if (v01) v = fmaf(w01, r[i01], v);
          if (v10) v = fmaf(w10, r[i10], v);
          if (v11) v = fmaf(w11, r[i11], v);
          Bt[pl * 40 + cb + j] = (_Float16)(v * m);
        }
      }
      __syncthreads();

      // A tile from global (L2-resident, 144KB total), B from LDS
      const _Float16* ap = wregh + ((size_t)kk * COUTC + (m0 + ml)) * CINC + c0 + sel * 16;
      v16h a = mk16(*(const v8h*)ap, *(const v8h*)(ap + 8));
      const int kb = sel * 16;
#pragma unroll
      for (int nt = 0; nt < 4; ++nt) {
        const _Float16* bp = &Bt[(nt * 16 + ml) * 40 + kb];
        v16h bb = mk16(*(const v8h*)bp, *(const v8h*)(bp + 8));
        acc[nt] = __builtin_amdgcn_wmma_f32_16x16x32_f16(false, a, false, bb, (short)0, acc[nt], false, false);
      }
      __syncthreads();  // protect Bt before next staging pass
    }
  }

  float* ob = out + (size_t)b * COUTC * HW;
#pragma unroll
  for (int nt = 0; nt < 4; ++nt) {
#pragma unroll
    for (int v = 0; v < 8; ++v) {
      int m = m0 + v + 8 * sel;
      ob[(size_t)m * HW + pix0 + nt * 16 + (lane & 15)] = acc[nt][v];
    }
  }
}

// ---------------------------------------------------------------------------
extern "C" void kernel_launch(void* const* d_in, const int* in_sizes, int n_in,
                              void* d_out, int out_size, void* d_ws, size_t ws_size,
                              hipStream_t stream) {
  const float* x     = (const float*)d_in[0];
  const float* pre_w = (const float*)d_in[1];
  const float* pre_b = (const float*)d_in[2];
  const float* off_w = (const float*)d_in[3];
  const float* off_b = (const float*)d_in[4];
  const float* mod_w = (const float*)d_in[5];
  const float* mod_b = (const float*)d_in[6];
  const float* reg_w = (const float*)d_in[7];

  char* ws = (char*)d_ws;
  _Float16* wpreh = (_Float16*)(ws);                         // 8192 B
  _Float16* wregh = (_Float16*)(ws + 8192);                  // 147456 B
  float* xp    = (float*)(ws + 155648);                      // 33.5 MB
  float* offsb = (float*)(ws + 155648 + 33554432);           // 9.44 MB
  float* maskb = (float*)(ws + 155648 + 33554432 + 9437184); // 4.72 MB
  float* out = (float*)d_out;

  k_prep<<<(CINC * CINC + KKC * COUTC * CINC + 255) / 256, 256, 0, stream>>>(
      pre_w, reg_w, wpreh, wregh);
  k_preconv<<<Bn * (HW / 64), 256, 0, stream>>>(x, wpreh, pre_b, xp);
  k_offmask<<<Bn * 64, 256, 0, stream>>>(xp, off_w, off_b, mod_w, mod_b, offsb, maskb);
  k_dcn_main<<<Bn * (HW / 64), 256, 0, stream>>>(xp, offsb, maskb, wregh, out);
}
